// MaskedSelfAttention_4492535791851
// MI455X (gfx1250) — matrix operand order repaired and checked by reading, not compiled, not run on hardware
//
#include <hip/hip_runtime.h>

#define BDIM 2
#define SEQ 2048
#define DMODEL 2048
#define NHEAD 16
#define HDIM 128

typedef __attribute__((ext_vector_type(16))) __bf16 v16bf;
typedef __attribute__((ext_vector_type(8)))  float  v8f;

union FragU {
  v16bf v;
  uint4 q[2];
};

__device__ __forceinline__ unsigned short f32_to_bf16(float x) {
  unsigned int u = __float_as_uint(x);
  u += 0x7FFFu + ((u >> 16) & 1u);   // round to nearest even
  return (unsigned short)(u >> 16);
}

// ---- WMMA fragment loaders (bf16, 16x16x32) -------------------------------
// A (16 rows x 32 k), row-major source with leading dim `ld` (elements).
// Lane<16: row=lane,  k = {0..7, 16..23}; lane>=16: row=lane-16, k = {8..15, 24..31}.
__device__ __forceinline__ v16bf load_a_frag(const unsigned short* p, long ld) {
  const int lane = threadIdx.x & 31;
  const int m    = lane & 15;
  const int kb   = (lane < 16) ? 0 : 8;
  const unsigned short* q = p + (long)m * ld + kb;
  FragU f;
  f.q[0] = *(const uint4*)(q);        // k = kb .. kb+7
  f.q[1] = *(const uint4*)(q + 16);   // k = kb+16 .. kb+23
  return f.v;
}

// B (32 k x 16 cols). Source is stored as rows of the *column* index (i.e. B^T
// row-major, leading dim `ld`): lane<16: col=lane, k = {0..7, 16..23};
// lane>=16: col=lane-16, k = {8..15, 24..31}.
__device__ __forceinline__ v16bf load_b_frag(const unsigned short* p, long ld) {
  const int lane = threadIdx.x & 31;
  const int n    = lane & 15;
  const int kb   = (lane >> 4) << 3;  // 0 or 8
  const unsigned short* q = p + (long)n * ld + kb;
  FragU f;
  f.q[0] = *(const uint4*)(q);
  f.q[1] = *(const uint4*)(q + 16);
  return f.v;
}

__device__ __forceinline__ v8f wmma_bf16(v16bf a, v16bf b, v8f c) {
  return __builtin_amdgcn_wmma_f32_16x16x32_bf16(false, a, false, b, (short)0, c,
                                                 false, false);
}

// Half-wave (16 lane) reductions; C-layout rows live within a 16-lane half.
__device__ __forceinline__ float hmax16(float v) {
  v = fmaxf(v, __shfl_xor(v, 1));
  v = fmaxf(v, __shfl_xor(v, 2));
  v = fmaxf(v, __shfl_xor(v, 4));
  v = fmaxf(v, __shfl_xor(v, 8));
  return v;
}
__device__ __forceinline__ float hsum16(float v) {
  v += __shfl_xor(v, 1);
  v += __shfl_xor(v, 2);
  v += __shfl_xor(v, 4);
  v += __shfl_xor(v, 8);
  return v;
}

// ---- 1) fp32 -> bf16 elementwise ------------------------------------------
__global__ void cvt_f32_bf16(const float* __restrict__ in,
                             unsigned short* __restrict__ out, long n) {
  long i = ((long)blockIdx.x * blockDim.x + threadIdx.x) * 4;
  if (i + 3 < n) {
    float4 v = *(const float4*)(in + i);
    ushort4 o;
    o.x = f32_to_bf16(v.x); o.y = f32_to_bf16(v.y);
    o.z = f32_to_bf16(v.z); o.w = f32_to_bf16(v.w);
    *(ushort4*)(out + i) = o;
  }
}

// ---- 2) W[k][n] (f32) -> Wt[n][k] (bf16), 32x32 LDS tiles -----------------
__global__ void transpose_w(const float* __restrict__ W,
                            unsigned short* __restrict__ Wt) {
  __shared__ unsigned short tile[32][33];
  const int c0 = blockIdx.x * 32, r0 = blockIdx.y * 32;
  const int tx = threadIdx.x, ty = threadIdx.y;
#pragma unroll
  for (int i = 0; i < 32; i += 8)
    tile[ty + i][tx] = f32_to_bf16(W[(long)(r0 + ty + i) * DMODEL + c0 + tx]);
  __syncthreads();
#pragma unroll
  for (int i = 0; i < 32; i += 8)
    Wt[(long)(c0 + ty + i) * DMODEL + r0 + tx] = tile[tx][ty + i];
}

// ---- 4) V[b][s][h*HD+hd] -> Vt[b][h][hd][s] (bf16) ------------------------
__global__ void transpose_v(const unsigned short* __restrict__ V,
                            unsigned short* __restrict__ Vt) {
  __shared__ unsigned short tile[32][33];
  const int z = blockIdx.z, b = z >> 4, h = z & 15;
  const int d0 = blockIdx.x * 32;   // hd
  const int s0 = blockIdx.y * 32;   // s
  const int tx = threadIdx.x, ty = threadIdx.y;
  const unsigned short* src = V + ((long)b * SEQ) * DMODEL + h * HDIM;
#pragma unroll
  for (int i = 0; i < 32; i += 8)
    tile[ty + i][tx] = src[(long)(s0 + ty + i) * DMODEL + d0 + tx];
  __syncthreads();
  unsigned short* dst = Vt + (long)z * HDIM * SEQ;
#pragma unroll
  for (int i = 0; i < 32; i += 8)
    dst[(long)(d0 + ty + i) * SEQ + s0 + tx] = tile[tx][ty + i];
}

// ---- 3/6) GEMM: C[M,N] = A_bf16[M,K] * Wt_bf16[N,K]^T + bias --------------
// block = 256 thr (8 waves); wave -> 32 rows x 128 cols (2x8 wmma accums).
// 16 WMMAs per k-step against 4 A-loads + 16 B-loads: ~1.8x the FLOP/byte of a
// 16-row tile, which matters since the B tiles are only shared through L0/L2.
template <bool OUT_F32>
__global__ __launch_bounds__(256) void gemm_bias(
    const unsigned short* __restrict__ A, const unsigned short* __restrict__ Bt,
    const float* __restrict__ bias, void* __restrict__ Cout) {
  const int wave = threadIdx.x >> 5;
  const int lane = threadIdx.x & 31;
  const long row0 = (long)blockIdx.y * 256 + wave * 32;
  const long col0 = (long)blockIdx.x * 128;

  const v8f vzero = {0.f, 0.f, 0.f, 0.f, 0.f, 0.f, 0.f, 0.f};
  v8f acc0[8], acc1[8];
#pragma unroll
  for (int t = 0; t < 8; ++t) { acc0[t] = vzero; acc1[t] = vzero; }

  for (int k0 = 0; k0 < DMODEL; k0 += 32) {
    v16bf af0 = load_a_frag(A + row0 * DMODEL + k0, DMODEL);
    v16bf af1 = load_a_frag(A + (row0 + 16) * DMODEL + k0, DMODEL);
#pragma unroll
    for (int t = 0; t < 8; ++t) {
      v16bf bf = load_b_frag(Bt + (col0 + t * 16) * DMODEL + k0, DMODEL);
      acc0[t] = wmma_bf16(af0, bf, acc0[t]);
      acc1[t] = wmma_bf16(af1, bf, acc1[t]);
    }
  }
  const int halfoff = (lane >> 4) << 3;  // C layout: row = r + 8*(lane>=16)
  const int n = lane & 15;
#pragma unroll
  for (int t = 0; t < 8; ++t) {
    const float bv = bias[col0 + t * 16 + n];
#pragma unroll
    for (int r = 0; r < 8; ++r) {
      float v0 = acc0[t][r] + bv;
      float v1 = acc1[t][r] + bv;
      long idx0 = (row0 + halfoff + r) * DMODEL + col0 + t * 16 + n;
      long idx1 = idx0 + 16 * DMODEL;
      if (OUT_F32) {
        ((float*)Cout)[idx0] = v0;
        ((float*)Cout)[idx1] = v1;
      } else {
        ((unsigned short*)Cout)[idx0] = f32_to_bf16(v0);
        ((unsigned short*)Cout)[idx1] = f32_to_bf16(v1);
      }
    }
  }
}

// ---- 5) causal flash attention --------------------------------------------
// grid: (S/128, B*H); 8 waves/block; each wave owns 16 query rows.
__global__ __launch_bounds__(256) void attn_kernel(
    const unsigned short* __restrict__ Q, const unsigned short* __restrict__ Km,
    const unsigned short* __restrict__ Vt, unsigned short* __restrict__ Out) {
  __shared__ __align__(16) unsigned short pbuf[8][512];  // per-wave P tile 16x32
  const int wave = threadIdx.x >> 5;
  const int lane = threadIdx.x & 31;
  const int bh = blockIdx.y;
  const int b = bh >> 4, h = bh & 15;
  const int q0 = blockIdx.x * 128 + wave * 16;

  const unsigned short* Qb = Q + ((long)b * SEQ) * DMODEL + h * HDIM;
  const unsigned short* Kb = Km + ((long)b * SEQ) * DMODEL + h * HDIM;
  const unsigned short* Vb = Vt + (long)bh * HDIM * SEQ;
  unsigned short* pw = &pbuf[wave][0];

  v16bf qf[4];  // full 16 x 128 Q tile kept in registers
#pragma unroll
  for (int c = 0; c < 4; ++c)
    qf[c] = load_a_frag(Qb + (long)q0 * DMODEL + c * 32, DMODEL);

  const v8f vzero = {0.f, 0.f, 0.f, 0.f, 0.f, 0.f, 0.f, 0.f};
  v8f acc[8];
  float rm[8], rl[8];
#pragma unroll
  for (int t = 0; t < 8; ++t) acc[t] = vzero;
#pragma unroll
  for (int r = 0; r < 8; ++r) { rm[r] = -1e30f; rl[r] = 0.f; }

  const int halfoff = (lane >> 4) << 3;
  const int col = lane & 15;
  const float scale = 0.02209708691207961f;  // 1/sqrt(D) per reference

  for (int kc = 0; kc < q0 + 16; kc += 32) {
    v8f c0 = vzero, c1 = vzero;
#pragma unroll
    for (int c = 0; c < 4; ++c) {  // contract over hd = 128
      v16bf kf0 = load_b_frag(Kb + (long)kc * DMODEL + c * 32, DMODEL);
      c0 = wmma_bf16(qf[c], kf0, c0);
      v16bf kf1 = load_b_frag(Kb + (long)(kc + 16) * DMODEL + c * 32, DMODEL);
      c1 = wmma_bf16(qf[c], kf1, c1);
    }
    const bool need_mask = (kc + 31 > q0);  // wave-uniform
#pragma unroll
    for (int r = 0; r < 8; ++r) {
      float s0 = c0[r] * scale;
      float s1 = c1[r] * scale;
      const int qr = q0 + halfoff + r;
      if (need_mask) {
        if (kc + col > qr) s0 = -1e30f;
        if (kc + 16 + col > qr) s1 = -1e30f;
      }
      float mx   = hmax16(fmaxf(s0, s1));
      float mnew = fmaxf(rm[r], mx);
      float corr = __expf(rm[r] - mnew);
      rm[r] = mnew;
      float p0 = __expf(s0 - mnew);
      float p1 = __expf(s1 - mnew);
      rl[r] = rl[r] * corr + hsum16(p0 + p1);
#pragma unroll
      for (int t = 0; t < 8; ++t) acc[t][r] *= corr;
      pw[(halfoff + r) * 32 + col]      = f32_to_bf16(p0);
      pw[(halfoff + r) * 32 + 16 + col] = f32_to_bf16(p1);
    }
    asm volatile("s_wait_dscnt 0x0" ::: "memory");  // per-wave LDS RAW guard
    v16bf pf = load_a_frag(pw, 32);
#pragma unroll
    for (int t = 0; t < 8; ++t) {  // O += P * V over 8 hd tiles
      v16bf vf = load_b_frag(Vb + (long)(t * 16) * SEQ + kc, SEQ);
      acc[t] = wmma_bf16(pf, vf, acc[t]);
    }
  }
#pragma unroll
  for (int t = 0; t < 8; ++t) {
#pragma unroll
    for (int r = 0; r < 8; ++r) {
      float o = acc[t][r] / rl[r];
      long row = (long)b * SEQ + q0 + halfoff + r;
      Out[row * DMODEL + h * HDIM + t * 16 + col] = f32_to_bf16(o);
    }
  }
}

// ---------------------------------------------------------------------------
extern "C" void kernel_launch(void* const* d_in, const int* in_sizes, int n_in,
                              void* d_out, int out_size, void* d_ws, size_t ws_size,
                              hipStream_t stream) {
  (void)in_sizes; (void)n_in; (void)out_size; (void)ws_size;
  const float* x  = (const float*)d_in[0];
  const float* Wq = (const float*)d_in[1];
  const float* bq = (const float*)d_in[2];
  const float* Wk = (const float*)d_in[3];
  const float* bk = (const float*)d_in[4];
  const float* Wv = (const float*)d_in[5];
  const float* bv = (const float*)d_in[6];
  const float* Wp = (const float*)d_in[7];
  const float* bp = (const float*)d_in[8];

  char* ws = (char*)d_ws;
  const size_t MB = 1024u * 1024u;
  // layout (112 MB): xb doubles as attention output (x is dead by then)
  unsigned short* xb  = (unsigned short*)(ws);             // 16 MB  x / attn-out bf16
  unsigned short* Wqt = (unsigned short*)(ws + 16 * MB);   //  8 MB
  unsigned short* Wkt = (unsigned short*)(ws + 24 * MB);
  unsigned short* Wvt = (unsigned short*)(ws + 32 * MB);
  unsigned short* Wpt = (unsigned short*)(ws + 40 * MB);
  unsigned short* Qb  = (unsigned short*)(ws + 48 * MB);   // 16 MB
  unsigned short* Kb  = (unsigned short*)(ws + 64 * MB);
  unsigned short* Vb  = (unsigned short*)(ws + 80 * MB);
  unsigned short* Vt  = (unsigned short*)(ws + 96 * MB);

  const long ntok = (long)BDIM * SEQ;  // 4096

  // 1) x -> bf16
  cvt_f32_bf16<<<dim3((unsigned)((ntok * DMODEL) / (256 * 4))), dim3(256), 0,
                 stream>>>(x, xb, ntok * DMODEL);

  // 2) weight transposes (f32 -> bf16, W^T)
  dim3 tb(32, 8);
  dim3 tgw(DMODEL / 32, DMODEL / 32);
  transpose_w<<<tgw, tb, 0, stream>>>(Wq, Wqt);
  transpose_w<<<tgw, tb, 0, stream>>>(Wk, Wkt);
  transpose_w<<<tgw, tb, 0, stream>>>(Wv, Wvt);
  transpose_w<<<tgw, tb, 0, stream>>>(Wp, Wpt);

  // 3) Q/K/V projections (bf16 out); block tile = 256 rows x 128 cols
  dim3 gb(256);
  dim3 gg(DMODEL / 128, (unsigned)(ntok / 256));
  gemm_bias<false><<<gg, gb, 0, stream>>>(xb, Wqt, bq, Qb);
  gemm_bias<false><<<gg, gb, 0, stream>>>(xb, Wkt, bk, Kb);
  gemm_bias<false><<<gg, gb, 0, stream>>>(xb, Wvt, bv, Vb);

  // 4) per-head V transpose
  dim3 tgv(HDIM / 32, SEQ / 32, BDIM * NHEAD);
  transpose_v<<<tgv, tb, 0, stream>>>(Vb, Vt);

  // 5) causal flash attention (writes bf16 into xb)
  dim3 ag(SEQ / 128, BDIM * NHEAD);
  attn_kernel<<<ag, dim3(256), 0, stream>>>(Qb, Kb, Vt, xb);

  // 6) output projection (f32 out)
  gemm_bias<true><<<gg, gb, 0, stream>>>(xb, Wpt, bp, d_out);
}
